// ExpertsChooseMlp_71760313581580
// MI455X (gfx1250) — matrix operand order repaired and checked by reading, not compiled, not run on hardware
//
#include <hip/hip_runtime.h>
#include <math.h>

// ---------------------------------------------------------------------------
// MoE experts-choose MLP for MI455X (gfx1250, wave32, WMMA).
// Four einsums -> one templated batched bf16-WMMA GEMM:
//   Out[m,n] = sum_k A(k,m) * B(k,n)  (+bias, +gelu), f32 accumulate.
// All strides are compile-time template constants so staging address math
// folds to immediates; global traffic uses b128 loads (memory-bound problem:
// ~0.4 GB total => ~17us floor at 23.3 TB/s).
// ---------------------------------------------------------------------------

typedef __bf16 bf16_t;
typedef __attribute__((ext_vector_type(16))) __bf16 v16bf;
typedef __attribute__((ext_vector_type(8)))  __bf16 v8bf;
typedef __attribute__((ext_vector_type(4)))  __bf16 v4bf;
typedef __attribute__((ext_vector_type(8)))  float  v8f;

#define TILE_M 128
#define TILE_N 128
#define TILE_K 64
#define LDS_K  72   // 64 + 8 pad: row stride 144 B, multiple of 16 B

__device__ __forceinline__ float gelu_erf(float v) {
    return 0.5f * v * (1.0f + erff(v * 0.70710678118654752f));
}

// 256 threads = 8 waves (wave32). Wave grid 4(M) x 2(N): each wave owns 32x64.
template <int KDIM,
          long long SAK, long long SAM,        // A(k,m) element strides
          long long SBK,                        // B(k,n): n-stride is 1 (all cases)
          long long LDO,                        // Out row stride
          long long SAB, long long SAE,         // A batch (b,e) offsets
          long long SBB, long long SBE,         // B batch (b,e) offsets
          long long SOB, long long SOE,         // Out batch (b,e) offsets
          long long SBIASE,                     // bias per-expert stride
          int EDIV, bool GELU, bool HAS_BIAS>
__global__ __launch_bounds__(256)
void moe_wmma_gemm(const float* __restrict__ A,
                   const float* __restrict__ Bm,
                   const float* __restrict__ bias,
                   float* __restrict__ Out) {
    static_assert(SAK == 1 || SAM == 1, "one A stride must be unit");

    __shared__ bf16_t sA[TILE_M][LDS_K];   // [m][k]  fragment-major for A
    __shared__ bf16_t sB[TILE_N][LDS_K];   // [n][k]  fragment-major for B

    const int tid  = threadIdx.x;
    const int lane = tid & 31;
    const int wave = tid >> 5;
    const int wm   = wave & 3;   // M offset wm*32
    const int wn   = wave >> 2;  // N offset wn*64

    const int zb = (int)blockIdx.z / EDIV;
    const int ze = (int)blockIdx.z % EDIV;
    const long long offA = (long long)zb * SAB + (long long)ze * SAE;
    const long long offB = (long long)zb * SBB + (long long)ze * SBE;
    const long long offO = (long long)zb * SOB + (long long)ze * SOE;
    const long long m0 = (long long)blockIdx.y * TILE_M;
    const long long n0 = (long long)blockIdx.x * TILE_N;

    const int halfSel = lane >> 4;         // 0: lanes 0-15, 1: lanes 16-31
    const int mn      = lane & 15;
    const int kbaseA  = halfSel ? 8 : 0;   // A frag: K in {kb..kb+7, kb+16..kb+23}
    const int kbaseB  = halfSel ? 8 : 0;   // B frag: K in {kb..kb+7, kb+16..kb+23}

    v8f acc[2][4];
#pragma unroll
    for (int mi = 0; mi < 2; ++mi)
#pragma unroll
        for (int ni = 0; ni < 4; ++ni)
#pragma unroll
            for (int r = 0; r < 8; ++r) acc[mi][ni][r] = 0.0f;

    for (int k0 = 0; k0 < KDIM; k0 += TILE_K) {
        // ---- stage A tile (TILE_M x TILE_K) as bf16, b128 global loads ----
#pragma unroll 2
        for (int i = tid; i < TILE_M * TILE_K / 4; i += 256) {
            if constexpr (SAK == 1) {            // k is the unit-stride dim
                const int k4 = (i & (TILE_K / 4 - 1)) * 4;
                const int m  = i >> 4;           // log2(TILE_K/4)
                const float4 v =
                    *(const float4*)&A[offA + (m0 + m) * SAM + (k0 + k4)];
                v4bf p = { (bf16_t)v.x, (bf16_t)v.y, (bf16_t)v.z, (bf16_t)v.w };
                *(v4bf*)&sA[m][k4] = p;          // one b64 LDS store
            } else {                              // m is the unit-stride dim
                const int m4 = (i & (TILE_M / 4 - 1)) * 4;
                const int k  = i >> 5;           // log2(TILE_M/4)
                const float4 v =
                    *(const float4*)&A[offA + (long long)(k0 + k) * SAK + (m0 + m4)];
                sA[m4 + 0][k] = (bf16_t)v.x;
                sA[m4 + 1][k] = (bf16_t)v.y;
                sA[m4 + 2][k] = (bf16_t)v.z;
                sA[m4 + 3][k] = (bf16_t)v.w;
            }
        }
        // ---- stage B tile (TILE_K x TILE_N), n is always unit-stride ----
#pragma unroll 2
        for (int i = tid; i < TILE_N * TILE_K / 4; i += 256) {
            const int n4 = (i & (TILE_N / 4 - 1)) * 4;
            const int k  = i >> 5;
            const float4 v =
                *(const float4*)&Bm[offB + (long long)(k0 + k) * SBK + (n0 + n4)];
            sB[n4 + 0][k] = (bf16_t)v.x;
            sB[n4 + 1][k] = (bf16_t)v.y;
            sB[n4 + 2][k] = (bf16_t)v.z;
            sB[n4 + 3][k] = (bf16_t)v.w;
        }
        __syncthreads();

        // ---- 2 K-steps of 32: b128 LDS fragment loads + 8 WMMAs each ----
#pragma unroll
        for (int kk = 0; kk < TILE_K; kk += 32) {
            v16bf afr[2], bfr[4];
#pragma unroll
            for (int mi = 0; mi < 2; ++mi) {
                const int m = wm * 32 + mi * 16 + mn;
                v8bf lo = *(const v8bf*)&sA[m][kk + kbaseA];
                v8bf hi = *(const v8bf*)&sA[m][kk + kbaseA + 16];
                afr[mi] = __builtin_shufflevector(lo, hi, 0, 1, 2, 3, 4, 5, 6, 7,
                                                  8, 9, 10, 11, 12, 13, 14, 15);
            }
#pragma unroll
            for (int ni = 0; ni < 4; ++ni) {
                const int n = wn * 64 + ni * 16 + mn;
                v8bf lo = *(const v8bf*)&sB[n][kk + kbaseB];
                v8bf hi = *(const v8bf*)&sB[n][kk + kbaseB + 16];
                bfr[ni] = __builtin_shufflevector(lo, hi, 0, 1, 2, 3, 4, 5, 6, 7,
                                                  8, 9, 10, 11, 12, 13, 14, 15);
            }
#pragma unroll
            for (int mi = 0; mi < 2; ++mi)
#pragma unroll
                for (int ni = 0; ni < 4; ++ni)
                    acc[mi][ni] = __builtin_amdgcn_wmma_f32_16x16x32_bf16(
                        false, afr[mi], false, bfr[ni],
                        (short)0, acc[mi][ni], false, false);
        }
        __syncthreads();
    }

    // ---- epilogue: optional per-n bias + exact-erf GELU, coalesced stores ----
    const float* biasE = HAS_BIAS ? (bias + (long long)ze * SBIASE) : nullptr;
#pragma unroll
    for (int mi = 0; mi < 2; ++mi) {
#pragma unroll
        for (int ni = 0; ni < 4; ++ni) {
            const long long ng = n0 + wn * 64 + ni * 16 + mn;
            float bn = HAS_BIAS ? biasE[ng] : 0.0f;
#pragma unroll
            for (int r = 0; r < 8; ++r) {
                const long long mg = m0 + wm * 32 + mi * 16 + r + (halfSel << 3);
                float v = acc[mi][ni][r] + bn;
                if (GELU) v = gelu_erf(v);
                Out[offO + mg * LDO + ng] = v;
            }
        }
    }
}

// ---------------------------------------------------------------------------
// Problem constants
// ---------------------------------------------------------------------------
constexpr int  Bc = 4, T = 2048, D = 512, E = 4, O = 512, C = 1024, HE = 512;
constexpr long long EC = (long long)E * C;

extern "C" void kernel_launch(void* const* d_in, const int* in_sizes, int n_in,
                              void* d_out, int out_size, void* d_ws, size_t ws_size,
                              hipStream_t stream) {
    (void)in_sizes; (void)n_in; (void)out_size; (void)ws_size;

    const float* x    = (const float*)d_in[0];
    const float* disp = (const float*)d_in[1];
    const float* comb = (const float*)d_in[2];
    const float* w1   = (const float*)d_in[3];
    const float* b1   = (const float*)d_in[4];
    const float* w2   = (const float*)d_in[5];
    const float* b2   = (const float*)d_in[6];
    float* out = (float*)d_out;

    float* xd = (float*)d_ws;                      // [B,E,C,D]
    float* h  = xd + (size_t)Bc * E * C * D;       // [B,E,C,HE]
    float* y  = h  + (size_t)Bc * E * C * HE;      // [B,E,C,O]

    dim3 blk(256);

    // 1) xd[b,e,c,d] = sum_t disp[b,t,e,c] * x[b,t,d]   (M=C, N=D, K=T)
    moe_wmma_gemm<T, EC, 1, D, D,
                  (long long)T * EC, C, (long long)T * D, 0,
                  EC * D, (long long)C * D, 0, E, false, false>
        <<<dim3(D / TILE_N, C / TILE_M, Bc * E), blk, 0, stream>>>(
            disp, x, nullptr, xd);

    // 2) h = gelu(xd @ w1[e] + b1[e])                    (M=C, N=HE, K=D)
    moe_wmma_gemm<D, 1, D, HE, HE,
                  EC * D, (long long)C * D, 0, (long long)D * HE,
                  EC * HE, (long long)C * HE, HE, E, true, true>
        <<<dim3(HE / TILE_N, C / TILE_M, Bc * E), blk, 0, stream>>>(
            xd, w1, b1, h);

    // 3) y = h @ w2[e]                                   (M=C, N=O, K=HE)
    moe_wmma_gemm<HE, 1, HE, O, O,
                  EC * HE, (long long)C * HE, 0, (long long)HE * O,
                  EC * O, (long long)C * O, 0, E, false, false>
        <<<dim3(O / TILE_N, C / TILE_M, Bc * E), blk, 0, stream>>>(
            h, w2, nullptr, y);

    // 4) out[b,t,o] = sum_{ec} comb[b,t,ec] * y[b,ec,o] + b2[o]  (M=T,N=O,K=EC)
    moe_wmma_gemm<(int)EC, 1, EC, O, O,
                  (long long)T * EC, 0, EC * O, 0,
                  (long long)T * O, 0, 0, 1, false, true>
        <<<dim3(O / TILE_N, T / TILE_M, Bc), blk, 0, stream>>>(
            comb, y, b2, out);
}
